// Transformer_31421980737739
// MI455X (gfx1250) — matrix operand order repaired and checked by reading, not compiled, not run on hardware
//
#include <hip/hip_runtime.h>
#include <hip/hip_bf16.h>
#include <math.h>

typedef __bf16 bf16;
typedef bf16  v16bf __attribute__((ext_vector_type(16)));
typedef float v8f   __attribute__((ext_vector_type(8)));
typedef int   vi4   __attribute__((vector_size(16)));

#define DIMC   1024
#define DEPTHC 6
#define HEADSC 16
#define MLPC   4096
#define TOKC   4096
#define EPSC   1e-5f

// ---- gfx1250 async global->LDS copy (ASYNCcnt path), guarded -------------
#if defined(__has_builtin)
#if __has_builtin(__builtin_amdgcn_global_load_async_to_lds_b128)
#define ASYNC_LDS 1
#endif
#endif

__device__ __forceinline__ void ld_async_b128(const void* g, void* l) {
#ifdef ASYNC_LDS
    __builtin_amdgcn_global_load_async_to_lds_b128(
        (__attribute__((address_space(1))) vi4*)g,
        (__attribute__((address_space(3))) vi4*)l, 0, 0);
#else
    *reinterpret_cast<uint4*>(l) = *reinterpret_cast<const uint4*>(g);
#endif
}

__device__ __forceinline__ void wait_async() {
#ifdef ASYNC_LDS
#if __has_builtin(__builtin_amdgcn_s_wait_asynccnt)
    __builtin_amdgcn_s_wait_asynccnt(0);
#else
    asm volatile("s_wait_asynccnt 0" ::: "memory");
#endif
#endif
}

__device__ __forceinline__ bf16 f2bf(float f) {
    unsigned u = __builtin_bit_cast(unsigned, f);
    unsigned r = u + 0x7fffu + ((u >> 16) & 1u);  // round-to-nearest-even
    unsigned short h = (unsigned short)(r >> 16);
    return __builtin_bit_cast(bf16, h);
}

__device__ __forceinline__ v8f vzero8() {
    v8f v;
#pragma unroll
    for (int i = 0; i < 8; ++i) v[i] = 0.0f;
    return v;
}

// ---- WMMA fragment loaders (CDNA5 ISA 7.12.2 layouts, wave32) -------------
// A fragment: 16x32 bf16 tile, row-major, leading dim `ld`.
// lane<16: row=lane, K={0..7,16..23}; lane>=16: same rows, K={8..15,24..31}.
__device__ __forceinline__ v16bf frag_a(const bf16* __restrict__ t, int ld) {
    int lane = threadIdx.x & 31;
    int row  = lane & 15;
    int klo  = (lane >> 4) * 8;
    const bf16* p = t + row * ld + klo;
    v16bf a;
#pragma unroll
    for (int j = 0; j < 8; ++j) a[j] = p[j];
#pragma unroll
    for (int j = 0; j < 8; ++j) a[8 + j] = p[16 + j];
    return a;
}

// B fragment from B^T (NxK) tile: lane=n (lane&15); K={0..7,16..23} / {8..15,24..31} by lane half.
__device__ __forceinline__ v16bf frag_b(const bf16* __restrict__ t, int ld) {
    int lane = threadIdx.x & 31;
    int col  = lane & 15;
    int ks   = (lane >> 4) * 8;
    const bf16* p = t + col * ld + ks;
    v16bf b;
#pragma unroll
    for (int j = 0; j < 8; ++j) b[j] = p[j];
#pragma unroll
    for (int j = 0; j < 8; ++j) b[8 + j] = p[16 + j];
    return b;
}

__device__ __forceinline__ v8f wmma_bf16(v16bf a, v16bf b, v8f c) {
    return __builtin_amdgcn_wmma_f32_16x16x32_bf16(false, a, false, b,
                                                   (short)0, c, false, false);
}

// ---- LayerNorm: one wave per token (d=1024 -> 32 f32/lane), bf16 output ---
__global__ __launch_bounds__(256) void ln_kernel(const float* __restrict__ x,
                                                 const float* __restrict__ w,
                                                 const float* __restrict__ b,
                                                 bf16* __restrict__ y) {
    int wave = threadIdx.x >> 5, lane = threadIdx.x & 31;
    int tok  = blockIdx.x * 8 + wave;
    const float* xr = x + (size_t)tok * DIMC;
    float v[32];
    float s = 0.0f;
#pragma unroll
    for (int j = 0; j < 32; ++j) { v[j] = xr[lane + 32 * j]; s += v[j]; }
#pragma unroll
    for (int m = 1; m < 32; m <<= 1) s += __shfl_xor(s, m, 32);
    float mu = s * (1.0f / DIMC);
    float q = 0.0f;
#pragma unroll
    for (int j = 0; j < 32; ++j) { float d = v[j] - mu; q += d * d; }
#pragma unroll
    for (int m = 1; m < 32; m <<= 1) q += __shfl_xor(q, m, 32);
    float rs = rsqrtf(q * (1.0f / DIMC) + EPSC);
    bf16* yr = y + (size_t)tok * DIMC;
#pragma unroll
    for (int j = 0; j < 32; ++j) {
        int c = lane + 32 * j;
        yr[c] = f2bf((v[j] - mu) * rs * w[c] + b[c]);
    }
}

// ---- GEMM: C[M,N] = A_bf16[M,K] @ W_f32[K,N] + bias; optional GELU /
// f32 residual / bf16 or f32 output.
// Block tile 128x128, 8 waves as 4(M)x2(N); each wave 32x64 = 8 WMMA/K-step.
template <bool GELU, bool RESID, bool OUTBF>
__global__ __launch_bounds__(256) void gemm_kernel(
    const bf16* __restrict__ A, const float* __restrict__ W,
    const float* __restrict__ bias, const float* __restrict__ resid,
    float* __restrict__ outf, bf16* __restrict__ outb, int M, int N, int K) {
    __shared__ __align__(16) bf16 As[128 * 32];
    __shared__ __align__(16) bf16 Bt[128 * 32];
    int tid   = threadIdx.x;
    int wave  = tid >> 5;
    int waveM = wave >> 1, waveN = wave & 1;
    int m0 = blockIdx.y * 128, n0 = blockIdx.x * 128;

    v8f acc[2][4];
#pragma unroll
    for (int i = 0; i < 2; ++i)
#pragma unroll
        for (int j = 0; j < 4; ++j) acc[i][j] = vzero8();

    for (int k0 = 0; k0 < K; k0 += 32) {
        if (k0 + 32 < K) {  // gfx1250 global_prefetch_b8 on next tiles
            __builtin_prefetch(A + (size_t)(m0 + (tid >> 1)) * K + k0 + 32, 0, 1);
            __builtin_prefetch(W + (size_t)(k0 + 32 + (tid >> 3)) * N + n0, 0, 1);
        }
        // A tile: 128x32 bf16 via async global->LDS b128 (ASYNCcnt)
#pragma unroll
        for (int p = 0; p < 2; ++p) {
            int linear = p * 2048 + tid * 8;
            int r = linear >> 5, c = linear & 31;
            ld_async_b128(A + (size_t)(m0 + r) * K + k0 + c, &As[r * 32 + c]);
        }
        // W tile: 32x128 f32 -> Bt (128x32) bf16, transposed store
        {
            int k = tid >> 3, ng = (tid & 7) * 16;
            const float* src = W + (size_t)(k0 + k) * N + n0 + ng;
#pragma unroll
            for (int i = 0; i < 16; ++i) Bt[(ng + i) * 32 + k] = f2bf(src[i]);
        }
        wait_async();
        __syncthreads();
        v16bf a0 = frag_a(As + (waveM * 32 + 0) * 32, 32);
        v16bf a1 = frag_a(As + (waveM * 32 + 16) * 32, 32);
#pragma unroll
        for (int j = 0; j < 4; ++j) {
            v16bf b = frag_b(Bt + (waveN * 64 + j * 16) * 32, 32);
            acc[0][j] = wmma_bf16(a0, b, acc[0][j]);
            acc[1][j] = wmma_bf16(a1, b, acc[1][j]);
        }
        __syncthreads();
    }

    int lane = tid & 31;
    int hl = lane >> 4, col = lane & 15;
#pragma unroll
    for (int ti = 0; ti < 2; ++ti)
#pragma unroll
        for (int tj = 0; tj < 4; ++tj)
#pragma unroll
            for (int i = 0; i < 8; ++i) {
                int row = m0 + waveM * 32 + ti * 16 + i + 8 * hl;
                int c   = n0 + waveN * 64 + tj * 16 + col;
                float v = acc[ti][tj][i] + bias[c];
                if (GELU) v = 0.5f * v * (1.0f + erff(v * 0.70710678118f));
                size_t idx = (size_t)row * N + c;
                if (RESID) v += resid[idx];
                if (OUTBF) outb[idx] = f2bf(v);
                else       outf[idx] = v;
            }
}

// ---- Flash attention: one block per (b,h), 128 q-rows, 8 waves x 16 rows.
// Streams 32-key chunks; S and O via WMMA; online softmax; scale = d^-0.5. --
__global__ __launch_bounds__(256) void attn_kernel(const bf16* __restrict__ qkv,
                                                   bf16* __restrict__ out) {
    __shared__ __align__(16) bf16 Qs[128 * 64];
    __shared__ __align__(16) bf16 Ks[32 * 64];
    __shared__ __align__(16) bf16 Vt[64 * 32];
    __shared__ __align__(16) bf16 Ps[8 * 16 * 32];
    int tid  = threadIdx.x;
    int wave = tid >> 5, lane = tid & 31;
    int hl = lane >> 4, col = lane & 15;
    int bh = blockIdx.x;
    int bb = bh >> 4, hh = bh & 15;
    int q0 = blockIdx.y * 128;

    // Q block -> LDS via async copy (rows within this batch's 1024 tokens)
#pragma unroll
    for (int p = 0; p < 4; ++p) {
        int linear = p * 2048 + tid * 8;
        int r = linear >> 6, c = linear & 63;
        ld_async_b128(qkv + (size_t)(bb * 1024 + q0 + r) * 3072 + hh * 64 + c,
                      &Qs[r * 64 + c]);
    }

    float mrow[8], lrow[8];
    v8f o[4];
#pragma unroll
    for (int i = 0; i < 8; ++i) { mrow[i] = -3.0e38f; lrow[i] = 0.0f; }
#pragma unroll
    for (int j = 0; j < 4; ++j) o[j] = vzero8();
    const float scale = 0.03125f;  // 1024^-0.5 (reference scales by dim**-0.5)

    for (int c0 = 0; c0 < 1024; c0 += 32) {
        __syncthreads();  // Ks/Vt consumers of previous chunk done
        {
            int r = tid >> 3, cg = (tid & 7) * 8;
            size_t base = (size_t)(bb * 1024 + c0 + r) * 3072 + hh * 64 + cg;
            ld_async_b128(qkv + base + 1024, &Ks[r * 64 + cg]);  // K rows
            union { uint4 u; bf16 h[8]; } vd;                    // V transpose
            vd.u = *reinterpret_cast<const uint4*>(qkv + base + 2048);
#pragma unroll
            for (int i = 0; i < 8; ++i) Vt[(cg + i) * 32 + r] = vd.h[i];
        }
        wait_async();  // covers Q (first iter) + this chunk's K
        __syncthreads();

        const bf16* qb = Qs + (wave * 16) * 64;
        v16bf aq0 = frag_a(qb, 64);       // Q, hd 0..31
        v16bf aq1 = frag_a(qb + 32, 64);  // Q, hd 32..63
        v8f s0 = vzero8(), s1 = vzero8();
        s0 = wmma_bf16(aq0, frag_b(Ks, 64), s0);            // keys 0..15
        s0 = wmma_bf16(aq1, frag_b(Ks + 32, 64), s0);
        s1 = wmma_bf16(aq0, frag_b(Ks + 16 * 64, 64), s1);  // keys 16..31
        s1 = wmma_bf16(aq1, frag_b(Ks + 16 * 64 + 32, 64), s1);

        bf16* ps = Ps + wave * 512;
#pragma unroll
        for (int i = 0; i < 8; ++i) {
            float v0 = s0[i] * scale, v1 = s1[i] * scale;
            float r = fmaxf(v0, v1);
#pragma unroll
            for (int mk = 1; mk < 16; mk <<= 1) r = fmaxf(r, __shfl_xor(r, mk, 32));
            float mn = fmaxf(mrow[i], r);
            float alpha = __expf(mrow[i] - mn);
            mrow[i] = mn;
            float p0 = __expf(v0 - mn), p1 = __expf(v1 - mn);
            float rs = p0 + p1;
#pragma unroll
            for (int mk = 1; mk < 16; mk <<= 1) rs += __shfl_xor(rs, mk, 32);
            lrow[i] = lrow[i] * alpha + rs;
#pragma unroll
            for (int j = 0; j < 4; ++j) o[j][i] = o[j][i] * alpha;
            int prow = i + 8 * hl;
            ps[prow * 32 + col]      = f2bf(p0);
            ps[prow * 32 + 16 + col] = f2bf(p1);
        }
        asm volatile("s_wait_dscnt 0" ::: "memory");  // P scratch RAW (per-wave)
        v16bf ap = frag_a(ps, 32);
#pragma unroll
        for (int j = 0; j < 4; ++j)
            o[j] = wmma_bf16(ap, frag_b(Vt + j * 16 * 32, 32), o[j]);
    }

#pragma unroll
    for (int i = 0; i < 8; ++i) {
        float inv = 1.0f / lrow[i];
        int t = bb * 1024 + q0 + wave * 16 + i + 8 * hl;
        bf16* orow = out + (size_t)t * 1024 + hh * 64;
#pragma unroll
        for (int j = 0; j < 4; ++j) orow[j * 16 + col] = f2bf(o[j][i] * inv);
    }
}

extern "C" void kernel_launch(void* const* d_in, const int* in_sizes, int n_in,
                              void* d_out, int out_size, void* d_ws, size_t ws_size,
                              hipStream_t stream) {
    (void)in_sizes; (void)n_in; (void)out_size; (void)ws_size;
    const float* x_in   = (const float*)d_in[0];
    const float* ln1_w  = (const float*)d_in[1];
    const float* ln1_b  = (const float*)d_in[2];
    const float* qkv_w  = (const float*)d_in[3];
    const float* qkv_b  = (const float*)d_in[4];
    const float* proj_w = (const float*)d_in[5];
    const float* proj_b = (const float*)d_in[6];
    const float* ln2_w  = (const float*)d_in[7];
    const float* ln2_b  = (const float*)d_in[8];
    const float* w1     = (const float*)d_in[9];
    const float* b1     = (const float*)d_in[10];
    const float* w2     = (const float*)d_in[11];
    const float* b2     = (const float*)d_in[12];

    float* x = (float*)d_out;               // running residual stream (f32)
    char*  ws = (char*)d_ws;
    bf16* ybf   = (bf16*)(ws);              //  8 MB: LN out, reused as attn out
    bf16* qkvbf = (bf16*)(ws + (size_t)(8u  << 20));  // 24 MB
    bf16* hbf   = (bf16*)(ws + (size_t)(32u << 20));  // 32 MB

    (void)hipMemcpyAsync(x, x_in, (size_t)TOKC * DIMC * sizeof(float),
                         hipMemcpyDeviceToDevice, stream);

    for (int l = 0; l < DEPTHC; ++l) {
        ln_kernel<<<TOKC / 8, 256, 0, stream>>>(x, ln1_w + l * DIMC,
                                                ln1_b + l * DIMC, ybf);
        gemm_kernel<false, false, true><<<dim3(3 * DIMC / 128, TOKC / 128), 256, 0, stream>>>(
            ybf, qkv_w + (size_t)l * DIMC * 3 * DIMC, qkv_b + l * 3 * DIMC,
            nullptr, nullptr, qkvbf, TOKC, 3 * DIMC, DIMC);
        attn_kernel<<<dim3(4 * HEADSC, 1024 / 128), 256, 0, stream>>>(qkvbf, ybf);
        gemm_kernel<false, true, false><<<dim3(DIMC / 128, TOKC / 128), 256, 0, stream>>>(
            ybf, proj_w + (size_t)l * DIMC * DIMC, proj_b + l * DIMC,
            x, x, nullptr, TOKC, DIMC, DIMC);
        ln_kernel<<<TOKC / 8, 256, 0, stream>>>(x, ln2_w + l * DIMC,
                                                ln2_b + l * DIMC, ybf);
        gemm_kernel<true, false, true><<<dim3(MLPC / 128, TOKC / 128), 256, 0, stream>>>(
            ybf, w1 + (size_t)l * DIMC * MLPC, b1 + l * MLPC,
            nullptr, nullptr, hbf, TOKC, MLPC, DIMC);
        gemm_kernel<false, true, false><<<dim3(DIMC / 128, TOKC / 128), 256, 0, stream>>>(
            hbf, w2 + (size_t)l * MLPC * DIMC, b2 + l * DIMC,
            x, x, nullptr, TOKC, DIMC, MLPC);
    }
}
